// BertSVDBlock_6786048327905
// MI455X (gfx1250) — matrix operand order repaired and checked by reading, not compiled, not run on hardware
//
#include <hip/hip_runtime.h>
#include <math.h>

// ---------------------------------------------------------------------------
// CDNA5 WMMA helpers (wave32, v_wmma_f32_16x16x32_bf16)
// ---------------------------------------------------------------------------
typedef __attribute__((ext_vector_type(16))) __bf16 v16bf;
typedef __attribute__((ext_vector_type(8)))  float  v8f;
typedef unsigned short ushort_t;

struct alignas(16) Chunk16 { unsigned u0, u1, u2, u3; };
struct alignas(16) F4 { float x[4]; };
struct alignas(8)  U2 { unsigned a, b; };
union Frag16 { Chunk16 c[2]; v16bf v; };

__device__ __forceinline__ ushort_t f2bf(float f) {
  union { float f; unsigned u; } x; x.f = f;
  unsigned r = x.u + 0x7FFFu + ((x.u >> 16) & 1u);   // round-to-nearest-even
  return (ushort_t)(r >> 16);
}
__device__ __forceinline__ unsigned pack2(float a, float b) {
  return (unsigned)f2bf(a) | ((unsigned)f2bf(b) << 16);
}
__device__ __forceinline__ void copy16(ushort_t* dst, const ushort_t* src) {
  *(Chunk16*)dst = *(const Chunk16*)src;             // b128 global->reg->LDS
}

// A fragment (16x32 bf16, ISA A-layout): lane L holds row (L&15);
// lanes 0-15: K {0..7, 16..23}; lanes 16-31: K {8..15, 24..31}.
__device__ __forceinline__ v16bf load_a_frag(const ushort_t* base, int stride, int lane) {
  const ushort_t* p = base + (lane & 15) * stride + ((lane >> 4) << 3);
  Frag16 f;
  f.c[0] = *(const Chunk16*)(p);
  f.c[1] = *(const Chunk16*)(p + 16);
  return f.v;
}
// B fragment (32x16 bf16) from transposed LDS tile Bt[n][k]:
// lane L holds column n=(L&15); lanes 0-15: K {0..7, 16..23}; lanes 16-31: K {8..15, 24..31}.
__device__ __forceinline__ v16bf load_b_frag(const ushort_t* base, int stride, int lane) {
  const ushort_t* p = base + (lane & 15) * stride + ((lane >> 4) << 3);
  Frag16 f;
  f.c[0] = *(const Chunk16*)(p);
  f.c[1] = *(const Chunk16*)(p + 16);
  return f.v;
}
__device__ __forceinline__ v8f wmma_bf16(v16bf a, v16bf b, v8f c) {
  return __builtin_amdgcn_wmma_f32_16x16x32_bf16(false, a, false, b,
                                                 (short)0, c, false, false);
}
#define ZERO8 {0.f,0.f,0.f,0.f,0.f,0.f,0.f,0.f}

// ---------------------------------------------------------------------------
// Optional Tensor Data Mover staging (gfx1250 TDM), D# per ISA 08 §8.
// This toolchain exposes the 6-arg builtin:
//   (uint32x4 g0, int32x8 g1, int32x4 g2, int32x4 g3, int32x8 g4, i32 cpol)
// ---------------------------------------------------------------------------
#if defined(__has_builtin)
#if __has_builtin(__builtin_amdgcn_tensor_load_to_lds) && \
    __has_builtin(__builtin_amdgcn_s_wait_tensorcnt)
#define HAVE_TDM 1
#endif
#endif

#ifdef HAVE_TDM
typedef __attribute__((ext_vector_type(4))) unsigned int u32x4;
typedef __attribute__((ext_vector_type(4))) int i32x4;
typedef __attribute__((ext_vector_type(8))) int i32x8;

__device__ __forceinline__ unsigned lds_off(const void* p) {
  return (unsigned)(unsigned long long)p;   // low 32 bits of flat LDS addr = LDS byte offset
}
// 2D bf16 tile load: tileW elems/row, tileH rows, row stride strideElems.
// LDS padding: padInterval encodes (1<<enc)*2 DWORDs per burst, padAmount enc = DWORDs-1.
__device__ __forceinline__ void tdm_load_2d_bf16(unsigned ldsByteOff, const void* gsrc,
                                                 unsigned tileW, unsigned tileH,
                                                 unsigned strideElems,
                                                 unsigned padInterval, unsigned padAmount) {
  unsigned long long ga = (unsigned long long)gsrc;
  u32x4 g0;
  g0[0] = 1u;                                               // count=1, user mode
  g0[1] = ldsByteOff;                                       // lds_addr
  g0[2] = (unsigned)ga;                                     // global_addr[31:0]
  g0[3] = (unsigned)((ga >> 32) & 0x01FFFFFFull) | 0x80000000u; // addr[56:32] | type=2
  i32x8 g1;
  g1[0] = (int)((1u << 16) | (1u << 20) |                   // data_size=2B, pad_enable
                (padInterval << 22) | (padAmount << 25));
  g1[1] = (int)((tileW & 0xFFFFu) << 16);                   // tensor_dim0[15:0]
  g1[2] = (int)(((tileW >> 16) & 0xFFFFu) | ((tileH & 0xFFFFu) << 16)); // td0 hi | td1 lo
  g1[3] = (int)(((tileH >> 16) & 0xFFFFu) | ((tileW & 0xFFFFu) << 16)); // td1 hi | tile_dim0
  g1[4] = (int)(tileH & 0xFFFFu);                           // tile_dim1 (tile_dim2=0)
  g1[5] = (int)strideElems;                                 // tensor_dim0_stride[31:0]
  g1[6] = 0;
  g1[7] = 0;
  i32x4 gz = {0, 0, 0, 0};
  i32x8 gz8 = {0, 0, 0, 0, 0, 0, 0, 0};
  __builtin_amdgcn_tensor_load_to_lds(g0, g1, gz, gz, gz8, 0);
}
#endif

// ---------------------------------------------------------------------------
// f32 -> bf16 plain convert (8 elems/thread)
// ---------------------------------------------------------------------------
__global__ void cvt_bf16(const float* __restrict__ in, ushort_t* __restrict__ out) {
  size_t i = ((size_t)blockIdx.x * 256 + threadIdx.x) * 8;
  F4 a = *(const F4*)&in[i];
  F4 b = *(const F4*)&in[i + 4];
  Chunk16 o;
  o.u0 = pack2(a.x[0], a.x[1]);
  o.u1 = pack2(a.x[2], a.x[3]);
  o.u2 = pack2(b.x[0], b.x[1]);
  o.u3 = pack2(b.x[2], b.x[3]);
  *(Chunk16*)&out[i] = o;
}

// ---------------------------------------------------------------------------
// f32 [batch][R][C] -> bf16 transposed [batch][C][R] (32x32 tiles, 256 thr)
// ---------------------------------------------------------------------------
__global__ void cvt_t_bf16(const float* __restrict__ in, ushort_t* __restrict__ out,
                           int R, int C) {
  const int z = blockIdx.z;
  in  += (size_t)z * R * C;
  out += (size_t)z * R * C;
  const int rbase = blockIdx.x * 32;
  const int cbase = blockIdx.y * 32;
  __shared__ ushort_t t[32][33];
  const int tid = threadIdx.x;
  {
    int rr = tid >> 3;
    int cq = (tid & 7) * 4;
    F4 v = *(const F4*)&in[(size_t)(rbase + rr) * C + cbase + cq];
#pragma unroll
    for (int i = 0; i < 4; ++i) t[rr][cq + i] = f2bf(v.x[i]);
  }
  __syncthreads();
  {
    int cc = tid >> 3;
    int rq = (tid & 7) * 4;
    U2 o;
    o.a = (unsigned)t[rq + 0][cc] | ((unsigned)t[rq + 1][cc] << 16);
    o.b = (unsigned)t[rq + 2][cc] | ((unsigned)t[rq + 3][cc] << 16);
    *(U2*)&out[(size_t)(cbase + cc) * R + rbase + rq] = o;
  }
}

// ---------------------------------------------------------------------------
// Generic bf16 WMMA GEMM: C[M,N] = A[M,K] * Bt[N,K]^T, 32x64 tile, 128 thr.
// EPI: 0 plain, 2 +bias +exact GELU, 3 +bias +f32 residual.  OBF: 1 = bf16 out
// ---------------------------------------------------------------------------
template <int EPI, int OBF>
__global__ void gemm32x64(const ushort_t* __restrict__ A, const ushort_t* __restrict__ Bt,
                          const float* __restrict__ bias, const float* __restrict__ res,
                          void* __restrict__ Cout, int N, int K) {
  __shared__ __align__(16) ushort_t Ash[32][40];
  __shared__ __align__(16) ushort_t Bsh[64][40];
  const int tid  = threadIdx.x;
  const int lane = tid & 31;
  const int w    = tid >> 5;
  const int wy   = w >> 1, wx = w & 1;
  const int mbase = blockIdx.x * 32;
  const int nbase = blockIdx.y * 64;

  // per-thread staging pointers (1 A chunk, 2 B chunks of 16B per k-step)
  const int ar = tid >> 2, aq = (tid & 3) * 8;
  const ushort_t* sA = A + (size_t)(mbase + ar) * K + aq;
  const ushort_t* sB0 = Bt + (size_t)(nbase + ar) * K + aq;
  const ushort_t* sB1 = Bt + (size_t)(nbase + 32 + ar) * K + aq;

  v8f acc0 = ZERO8, acc1 = ZERO8;

  for (int kt = 0; kt < K; kt += 32) {
    copy16(&Ash[ar][aq], sA + kt);
    copy16(&Bsh[ar][aq], sB0 + kt);
    copy16(&Bsh[32 + ar][aq], sB1 + kt);
    if (kt + 32 < K) {
      __builtin_prefetch(sA + kt + 32, 0, 0);
      __builtin_prefetch(sB0 + kt + 32, 0, 0);
      __builtin_prefetch(sB1 + kt + 32, 0, 0);
    }
    __syncthreads();
    v16bf a  = load_a_frag(&Ash[16 * wy][0], 40, lane);
    v16bf b0 = load_b_frag(&Bsh[32 * wx][0], 40, lane);
    v16bf b1 = load_b_frag(&Bsh[32 * wx + 16][0], 40, lane);
    acc0 = wmma_bf16(a, b0, acc0);
    acc1 = wmma_bf16(a, b1, acc1);
    __syncthreads();
  }

#pragma unroll
  for (int t = 0; t < 2; ++t) {
    v8f acc = t ? acc1 : acc0;
#pragma unroll
    for (int v = 0; v < 8; ++v) {
      int m = mbase + 16 * wy + v + 8 * (lane >> 4);
      int n = nbase + 32 * wx + 16 * t + (lane & 15);
      float val = acc[v];
      if (EPI >= 2) val += bias[n];
      if (EPI == 2) val = 0.5f * val * (1.0f + erff(val * 0.70710678118654752f));
      if (EPI == 3) val += res[(size_t)m * N + n];
      if (OBF) ((ushort_t*)Cout)[(size_t)m * N + n] = f2bf(val);
      else     ((float*)Cout)[(size_t)m * N + n] = val;
    }
  }
}

// ---------------------------------------------------------------------------
// Fused low-rank QKV projection (bf16 in / bf16 out, V written transposed)
// grid (256, 12, 3), block 128.
// ---------------------------------------------------------------------------
__global__ void qkv_proj(const ushort_t* __restrict__ xbf,
                         const ushort_t* __restrict__ PtQ, const ushort_t* __restrict__ WtQ, const float* __restrict__ bq,
                         const ushort_t* __restrict__ PtK, const ushort_t* __restrict__ WtK, const float* __restrict__ bk,
                         const ushort_t* __restrict__ PtV, const ushort_t* __restrict__ WtV, const float* __restrict__ bv,
                         ushort_t* __restrict__ Qo, ushort_t* __restrict__ Ko, ushort_t* __restrict__ Vto) {
  const int z = blockIdx.z;
  const ushort_t* Pt = (z == 0) ? PtQ : (z == 1) ? PtK : PtV;   // [H][32][768]
  const ushort_t* Wt = (z == 0) ? WtQ : (z == 1) ? WtK : WtV;   // [H][64][32]
  const float*    bb = (z == 0) ? bq  : (z == 1) ? bk  : bv;
  const int h     = blockIdx.y;
  const int mbase = blockIdx.x * 32;
  const int tid   = threadIdx.x;
  const int lane  = tid & 31;
  const int w     = tid >> 5;
  const int wy    = w >> 1, wx = w & 1;

  __shared__ __align__(16) ushort_t Ash[32][40];
  __shared__ __align__(16) ushort_t Psh[32][40];
  __shared__ __align__(16) ushort_t Tsh[32][40];
  __shared__ __align__(16) ushort_t Wsh[64][40];

  const int ar = tid >> 2, aq = (tid & 3) * 8;
  const ushort_t* sA = xbf + (size_t)(mbase + ar) * 768 + aq;
  const ushort_t* sP = Pt + ((size_t)h * 32 + ar) * 768 + aq;

  v8f acc = ZERO8;
  for (int kt = 0; kt < 768; kt += 32) {
    copy16(&Ash[ar][aq], sA + kt);
    copy16(&Psh[ar][aq], sP + kt);
    if (kt + 32 < 768) {
      __builtin_prefetch(sA + kt + 32, 0, 0);
      __builtin_prefetch(sP + kt + 32, 0, 0);
    }
    __syncthreads();
    v16bf a = load_a_frag(&Ash[16 * wy][0], 40, lane);
    v16bf b = load_b_frag(&Psh[16 * wx][0], 40, lane);
    acc = wmma_bf16(a, b, acc);
    __syncthreads();
  }

#pragma unroll
  for (int v = 0; v < 8; ++v)
    Tsh[16 * wy + v + 8 * (lane >> 4)][16 * wx + (lane & 15)] = f2bf(acc[v]);
#pragma unroll
  for (int j = 0; j < 2; ++j) {              // Wt[h]: 64 rows x 64B
    int id = tid + 128 * j;
    int r = id >> 2, q = (id & 3) * 8;
    copy16(&Wsh[r][q], Wt + ((size_t)h * 64 + r) * 32 + q);
  }
  __syncthreads();

  v16bf a2 = load_a_frag(&Tsh[16 * wy][0], 40, lane);
  v16bf w0 = load_b_frag(&Wsh[32 * wx][0], 40, lane);
  v16bf w1 = load_b_frag(&Wsh[32 * wx + 16][0], 40, lane);
  v8f o0 = ZERO8, o1 = ZERO8;
  o0 = wmma_bf16(a2, w0, o0);
  o1 = wmma_bf16(a2, w1, o1);

#pragma unroll
  for (int t = 0; t < 2; ++t) {
#pragma unroll
    for (int v = 0; v < 8; ++v) {
      int mloc  = 16 * wy + v + 8 * (lane >> 4);
      int dh    = 32 * wx + 16 * t + (lane & 15);
      int token = mbase + mloc;
      int bidx  = token >> 10;
      int mm    = token & 1023;
      int bh2   = bidx * 12 + h;
      float val = (t ? o1[v] : o0[v]) + bb[h * 64 + dh];
      ushort_t bfv = f2bf(val);
      if (z == 0)      Qo[(((size_t)bh2 * 1024 + mm) << 6) + dh] = bfv;
      else if (z == 1) Ko[(((size_t)bh2 * 1024 + mm) << 6) + dh] = bfv;
      else             Vto[((size_t)bh2 * 64 + dh) * 1024 + mm] = bfv;  // transposed
    }
  }
}

// ---------------------------------------------------------------------------
// Flash attention (bf16 Q/K, V pre-transposed), online softmax, bf16 out.
// grid (32, 96), block 128.  K/V tiles staged via TDM when available.
// ---------------------------------------------------------------------------
__global__ void attn_kernel(const ushort_t* __restrict__ Q, const ushort_t* __restrict__ Kb,
                            const ushort_t* __restrict__ Vt, const int* __restrict__ mask,
                            ushort_t* __restrict__ out) {
  const int bh = blockIdx.y;
  const int b = bh / 12, h = bh % 12;
  const int mbase = blockIdx.x * 32;
  const int tid = threadIdx.x, lane = tid & 31, w = tid >> 5;
  const int wy = w >> 1, wx = w & 1;

  __shared__ __align__(16) ushort_t Qsh[32][72];
  __shared__ __align__(16) ushort_t Ksh[32][72];   // [key][dh], padded 64->72
  __shared__ __align__(16) ushort_t Psh[32][40];
  __shared__ __align__(16) ushort_t Vsh[64][40];   // [dh][key], padded 32->40
  __shared__ float wmaxs[2][32], wsums[2][32];
  __shared__ float run_m[32], run_l[32], fac[32], mnew[32];

#pragma unroll
  for (int j = 0; j < 2; ++j) {
    int id = tid + 128 * j;
    int r = id >> 3, q = (id & 7) * 8;
    copy16(&Qsh[r][q], Q + ((size_t)bh * 1024 + mbase + r) * 64 + q);
  }
  if (tid < 32) { run_m[tid] = -3.0e38f; run_l[tid] = 0.0f; }

#ifdef HAVE_TDM
  const unsigned ldsK = lds_off(&Ksh[0][0]);
  const unsigned ldsV = lds_off(&Vsh[0][0]);
#endif

  v8f o0 = ZERO8, o1 = ZERO8;

  for (int kb = 0; kb < 32; ++kb) {
    __syncthreads();
#ifdef HAVE_TDM
    if (w == 0) {
      // K tile: 32 rows x 64 bf16, row 128B (32 DW) + 16B pad -> enc (4,3)
      tdm_load_2d_bf16(ldsK, Kb + ((size_t)bh * 1024 + kb * 32) * 64, 64, 32, 64, 4, 3);
      // V tile: 64 rows x 32 bf16 from [dh][1024], row 64B (16 DW) + 16B pad -> enc (3,3)
      tdm_load_2d_bf16(ldsV, Vt + (size_t)bh * 64 * 1024 + kb * 32, 32, 64, 1024, 3, 3);
      __builtin_amdgcn_s_wait_tensorcnt(0);
    }
#else
#pragma unroll
    for (int j = 0; j < 2; ++j) {
      int id = tid + 128 * j;
      int r = id >> 3, q = (id & 7) * 8;
      copy16(&Ksh[r][q], Kb + ((size_t)bh * 1024 + kb * 32 + r) * 64 + q);
      int rv = id >> 2, qv = (id & 3) * 8;
      copy16(&Vsh[rv][qv], Vt + ((size_t)bh * 64 + rv) * 1024 + kb * 32 + qv);
    }
#endif
    const int keyn = kb * 32 + 16 * wx + (lane & 15);
    const int mval = mask[b * 1024 + keyn];
    __syncthreads();

    v8f s = ZERO8;
#pragma unroll
    for (int s2 = 0; s2 < 2; ++s2) {
      v16bf a  = load_a_frag(&Qsh[16 * wy][32 * s2], 72, lane);
      v16bf bk = load_b_frag(&Ksh[16 * wx][32 * s2], 72, lane);
      s = wmma_bf16(a, bk, s);
    }
    float sv[8];
#pragma unroll
    for (int v = 0; v < 8; ++v) {
      float t = s[v] * 0.125f;
      sv[v] = (mval == 0) ? -1.0e9f : t;
    }
#pragma unroll
    for (int v = 0; v < 8; ++v) {
      float r = sv[v];
      r = fmaxf(r, __shfl_xor(r, 1, 16));
      r = fmaxf(r, __shfl_xor(r, 2, 16));
      r = fmaxf(r, __shfl_xor(r, 4, 16));
      r = fmaxf(r, __shfl_xor(r, 8, 16));
      if ((lane & 15) == 0) wmaxs[wx][16 * wy + v + 8 * (lane >> 4)] = r;
    }
    __syncthreads();
    if (tid < 32) {
      float bm = fmaxf(wmaxs[0][tid], wmaxs[1][tid]);
      float mo = run_m[tid];
      float mn = fmaxf(mo, bm);
      float f = __expf(mo - mn);
      run_l[tid] *= f;
      run_m[tid] = mn;
      fac[tid] = f;
      mnew[tid] = mn;
    }
    __syncthreads();
#pragma unroll
    for (int v = 0; v < 8; ++v) {
      int row = 16 * wy + v + 8 * (lane >> 4);
      float p = __expf(sv[v] - mnew[row]);
      Psh[row][16 * wx + (lane & 15)] = f2bf(p);
      float r = p;
      r += __shfl_xor(r, 1, 16);
      r += __shfl_xor(r, 2, 16);
      r += __shfl_xor(r, 4, 16);
      r += __shfl_xor(r, 8, 16);
      if ((lane & 15) == 0) wsums[wx][row] = r;
      float f = fac[row];
      o0[v] *= f;
      o1[v] *= f;
    }
    __syncthreads();
    if (tid < 32) run_l[tid] += wsums[0][tid] + wsums[1][tid];

    v16bf pa  = load_a_frag(&Psh[16 * wy][0], 40, lane);
    v16bf vb0 = load_b_frag(&Vsh[32 * wx][0], 40, lane);
    v16bf vb1 = load_b_frag(&Vsh[32 * wx + 16][0], 40, lane);
    o0 = wmma_bf16(pa, vb0, o0);
    o1 = wmma_bf16(pa, vb1, o1);
  }
  __syncthreads();

#pragma unroll
  for (int t = 0; t < 2; ++t) {
#pragma unroll
    for (int v = 0; v < 8; ++v) {
      int row = 16 * wy + v + 8 * (lane >> 4);
      float l = run_l[row];
      float val = (t ? o1[v] : o0[v]) / l;
      int m = mbase + row;
      int dh = 32 * wx + 16 * t + (lane & 15);
      out[((size_t)(b * 1024 + m)) * 768 + h * 64 + dh] = f2bf(val);
    }
  }
}

// ---------------------------------------------------------------------------
// LayerNorm over 768; optional secondary bf16 output.
// ---------------------------------------------------------------------------
__global__ void ln_kernel(const float* __restrict__ in, const float* __restrict__ g,
                          const float* __restrict__ bta, float* __restrict__ outf,
                          ushort_t* __restrict__ outbf) {
  const int row = blockIdx.x, tid = threadIdx.x;
  __shared__ float red[256];
  float v0 = in[(size_t)row * 768 + tid];
  float v1 = in[(size_t)row * 768 + tid + 256];
  float v2 = in[(size_t)row * 768 + tid + 512];
  red[tid] = v0 + v1 + v2;
  __syncthreads();
  for (int o = 128; o > 0; o >>= 1) { if (tid < o) red[tid] += red[tid + o]; __syncthreads(); }
  float mean = red[0] * (1.0f / 768.0f);
  __syncthreads();
  float d0 = v0 - mean, d1 = v1 - mean, d2 = v2 - mean;
  red[tid] = d0 * d0 + d1 * d1 + d2 * d2;
  __syncthreads();
  for (int o = 128; o > 0; o >>= 1) { if (tid < o) red[tid] += red[tid + o]; __syncthreads(); }
  float rstd = rsqrtf(red[0] * (1.0f / 768.0f) + 1e-12f);
  float r0 = d0 * rstd * g[tid] + bta[tid];
  float r1 = d1 * rstd * g[tid + 256] + bta[tid + 256];
  float r2 = d2 * rstd * g[tid + 512] + bta[tid + 512];
  outf[(size_t)row * 768 + tid]       = r0;
  outf[(size_t)row * 768 + tid + 256] = r1;
  outf[(size_t)row * 768 + tid + 512] = r2;
  if (outbf) {
    outbf[(size_t)row * 768 + tid]       = f2bf(r0);
    outbf[(size_t)row * 768 + tid + 256] = f2bf(r1);
    outbf[(size_t)row * 768 + tid + 512] = f2bf(r2);
  }
}

// ---------------------------------------------------------------------------
// Pipeline launcher
// ---------------------------------------------------------------------------
extern "C" void kernel_launch(void* const* d_in, const int* in_sizes, int n_in,
                              void* d_out, int out_size, void* d_ws, size_t ws_size,
                              hipStream_t stream) {
  (void)in_sizes; (void)n_in; (void)out_size; (void)ws_size;
  const float* x   = (const float*)d_in[0];
  const int*   msk = (const int*)d_in[1];
  const float* Pq  = (const float*)d_in[2];
  const float* Vq  = (const float*)d_in[3];
  const float* bq  = (const float*)d_in[4];
  const float* Pk  = (const float*)d_in[5];
  const float* Vk  = (const float*)d_in[6];
  const float* bk  = (const float*)d_in[7];
  const float* Pv  = (const float*)d_in[8];
  const float* Vv  = (const float*)d_in[9];
  const float* bv  = (const float*)d_in[10];
  const float* Uo  = (const float*)d_in[11];
  const float* Vo  = (const float*)d_in[12];
  const float* bo  = (const float*)d_in[13];
  const float* U1  = (const float*)d_in[14];
  const float* V1  = (const float*)d_in[15];
  const float* b1  = (const float*)d_in[16];
  const float* U2  = (const float*)d_in[17];
  const float* V2  = (const float*)d_in[18];
  const float* b2  = (const float*)d_in[19];
  const float* g1  = (const float*)d_in[20];
  const float* be1 = (const float*)d_in[21];
  const float* g2  = (const float*)d_in[22];
  const float* be2 = (const float*)d_in[23];
  float* out = (float*)d_out;
  char* p = (char*)d_ws;

  const size_t SB = 12582912;            // T*768 bf16 bytes
  ushort_t* Qbf   = (ushort_t*)(p);
  ushort_t* Kbf   = (ushort_t*)(p + SB);
  ushort_t* Vtbf  = (ushort_t*)(p + 2 * SB);
  ushort_t* AObf  = (ushort_t*)(p + 3 * SB);
  ushort_t* MIDA  = (ushort_t*)(p);      // [T,3072] bf16 overlays Q/K/Vt/AO (4*SB)
  ushort_t* xbf   = (ushort_t*)(p + 4 * SB);
  ushort_t* X1bf  = (ushort_t*)(p + 5 * SB);
  ushort_t* R256  = (ushort_t*)(p + 6 * SB);            // T*256 bf16 = 4 MiB
  float*    PRE   = (float*)(p + 6 * SB + 4194304);
  float*    X1f   = (float*)(p + 6 * SB + 4194304 + 25165824);
  char* wp = p + 6 * SB + 4194304 + 2 * 25165824;
  ushort_t* PtQ = (ushort_t*)(wp);            wp += 589824;   // [12][32][768]
  ushort_t* PtK = (ushort_t*)(wp);            wp += 589824;
  ushort_t* PtV = (ushort_t*)(wp);            wp += 589824;
  ushort_t* WtQ = (ushort_t*)(wp);            wp += 49152;    // [12][64][32]
  ushort_t* WtK = (ushort_t*)(wp);            wp += 49152;
  ushort_t* WtV = (ushort_t*)(wp);            wp += 49152;
  ushort_t* UoT = (ushort_t*)(wp);            wp += 393216;   // [256][768]
  ushort_t* VoT = (ushort_t*)(wp);            wp += 393216;   // [768][256]
  ushort_t* U1T = (ushort_t*)(wp);            wp += 393216;   // [256][768]
  ushort_t* V1T = (ushort_t*)(wp);            wp += 1572864;  // [3072][256]
  ushort_t* U2T = (ushort_t*)(wp);            wp += 1572864;  // [256][3072]
  ushort_t* V2T = (ushort_t*)(wp);            wp += 393216;   // [768][256]

  dim3 b128(128), b256(256);
  // one-time conversions (x plain; weights converted + transposed to [N][K])
  cvt_bf16<<<dim3(3072), b256, 0, stream>>>(x, xbf);
  cvt_t_bf16<<<dim3(24, 1, 12), b256, 0, stream>>>(Pq, PtQ, 768, 32);
  cvt_t_bf16<<<dim3(24, 1, 12), b256, 0, stream>>>(Pk, PtK, 768, 32);
  cvt_t_bf16<<<dim3(24, 1, 12), b256, 0, stream>>>(Pv, PtV, 768, 32);
  cvt_t_bf16<<<dim3(1, 2, 12), b256, 0, stream>>>(Vq, WtQ, 32, 64);
  cvt_t_bf16<<<dim3(1, 2, 12), b256, 0, stream>>>(Vk, WtK, 32, 64);
  cvt_t_bf16<<<dim3(1, 2, 12), b256, 0, stream>>>(Vv, WtV, 32, 64);
  cvt_t_bf16<<<dim3(24, 8, 1), b256, 0, stream>>>(Uo, UoT, 768, 256);
  cvt_t_bf16<<<dim3(8, 24, 1), b256, 0, stream>>>(Vo, VoT, 256, 768);
  cvt_t_bf16<<<dim3(24, 8, 1), b256, 0, stream>>>(U1, U1T, 768, 256);
  cvt_t_bf16<<<dim3(8, 96, 1), b256, 0, stream>>>(V1, V1T, 256, 3072);
  cvt_t_bf16<<<dim3(96, 8, 1), b256, 0, stream>>>(U2, U2T, 3072, 256);
  cvt_t_bf16<<<dim3(8, 24, 1), b256, 0, stream>>>(V2, V2T, 256, 768);

  qkv_proj<<<dim3(256, 12, 3), b128, 0, stream>>>(xbf, PtQ, WtQ, bq, PtK, WtK, bk,
                                                  PtV, WtV, bv, Qbf, Kbf, Vtbf);
  attn_kernel<<<dim3(32, 96), b128, 0, stream>>>(Qbf, Kbf, Vtbf, msk, AObf);
  gemm32x64<0, 1><<<dim3(256, 4),  b128, 0, stream>>>(AObf, UoT, nullptr, nullptr, R256, 256, 768);
  gemm32x64<3, 0><<<dim3(256, 12), b128, 0, stream>>>(R256, VoT, bo, x, PRE, 768, 256);
  ln_kernel<<<dim3(8192), b256, 0, stream>>>(PRE, g1, be1, X1f, X1bf);
  gemm32x64<0, 1><<<dim3(256, 4),  b128, 0, stream>>>(X1bf, U1T, nullptr, nullptr, R256, 256, 768);
  gemm32x64<2, 1><<<dim3(256, 48), b128, 0, stream>>>(R256, V1T, b1, nullptr, MIDA, 3072, 256);
  gemm32x64<0, 1><<<dim3(256, 4),  b128, 0, stream>>>(MIDA, U2T, nullptr, nullptr, R256, 256, 3072);
  gemm32x64<3, 0><<<dim3(256, 12), b128, 0, stream>>>(R256, V2T, b2, X1f, PRE, 768, 256);
  ln_kernel<<<dim3(8192), b256, 0, stream>>>(PRE, g2, be2, out, nullptr);
}
